// MemoryConsolidationLoop_24043226923370
// MI455X (gfx1250) — matrix operand order repaired and checked by reading, not compiled, not run on hardware
//
#include <hip/hip_runtime.h>
#include <math.h>

#define DIMX   2048
#define NSLOTS 256
#define TSTEPS 64
#define PI_F   3.14159265358979323846f

typedef __attribute__((ext_vector_type(16))) __bf16 v16bf;
typedef __attribute__((ext_vector_type(8)))  float  v8f;

__device__ __forceinline__ float sigf(float x) { return 1.0f / (1.0f + expf(-x)); }

// Pack 16 fp32 (as 4x float4, in K-order per the ISA's bf16 A/B VGPR layouts)
// into a 16-wide bf16 ext-vector (lowers to v_cvt_pk_bf16_f32 pairs).
__device__ __forceinline__ v16bf cvt16(float4 a0, float4 a1, float4 a2, float4 a3) {
    v16bf v;
    v[0]  = (__bf16)a0.x; v[1]  = (__bf16)a0.y; v[2]  = (__bf16)a0.z; v[3]  = (__bf16)a0.w;
    v[4]  = (__bf16)a1.x; v[5]  = (__bf16)a1.y; v[6]  = (__bf16)a1.z; v[7]  = (__bf16)a1.w;
    v[8]  = (__bf16)a2.x; v[9]  = (__bf16)a2.y; v[10] = (__bf16)a2.z; v[11] = (__bf16)a2.w;
    v[12] = (__bf16)a3.x; v[13] = (__bf16)a3.y; v[14] = (__bf16)a3.z; v[15] = (__bf16)a3.w;
    return v;
}

// ---------------------------------------------------------------------------
// Prep: noisy = traces[idx] + 0.01*noise ; voltage = sigmoid(cos(phase-pi)*vw+vb)
// grid (DIMX/256, TSTEPS), block 256
// ---------------------------------------------------------------------------
__global__ __launch_bounds__(256)
void prep_kernel(const float* __restrict__ traces, const float* __restrict__ noise,
                 const int* __restrict__ idx, const float* __restrict__ vw,
                 const float* __restrict__ vb,
                 float* __restrict__ noisy, float* __restrict__ voltage) {
    int t = blockIdx.y;
    int d = blockIdx.x * blockDim.x + threadIdx.x;
    float ph   = fmodf(2.0f * PI_F * 6.0f * 0.01f * (float)(t + 1), 2.0f * PI_F);
    float psig = cosf(ph - PI_F);
    long  row  = (long)idx[t];
    noisy[t * DIMX + d]   = traces[row * (long)DIMX + d] + 0.01f * noise[t * DIMX + d];
    voltage[t * DIMX + d] = sigf(psig * vw[d] + vb[d]);
}

// ---------------------------------------------------------------------------
// Fused split-K WMMA GEMM: Y = epilogue(X @ W^T + bias)
//   X:(64 x K) row-major, W:(Nout x K) row-major, Y:(64 x Nout)
// One 8-wave block per 16x16 output tile; wave w reduces K-chunk [w*K/8,(w+1)*K/8)
// with v_wmma_f32_16x16x32_bf16 (2 per unrolled iter), then a deterministic
// LDS tree-sum combines the 8 partial accumulators.
// MODE: 0=none, 1=relu, 2=sigmoid, 3=sigmoid(y)*aux[m,n]
// grid: 4 * (Nout/16) blocks of 256 threads.
// ---------------------------------------------------------------------------
template <int MODE>
__global__ __launch_bounds__(256)
void gemm_wmma_bf16(const float* __restrict__ X, const float* __restrict__ Wm,
                    const float* __restrict__ bias, const float* __restrict__ aux,
                    float* __restrict__ Y, int K, int Nout) {
    __shared__ float red[8][256];

    const int wid    = threadIdx.x >> 5;
    const int lane   = threadIdx.x & 31;
    const int l      = lane & 15;
    const int hi     = lane >> 4;          // half-wave select
    const int tilesN = Nout >> 4;
    const int tm     = blockIdx.x / tilesN;
    const int tn     = blockIdx.x % tilesN;

    const int kChunk = K >> 3;             // 256 for K=2048
    const int kBeg   = wid * kChunk;
    const int kEnd   = kBeg + kChunk;

    const float* xrow = X  + (size_t)(tm * 16 + l) * K;   // A: row l of tile
    const float* wrow = Wm + (size_t)(tn * 16 + l) * K;   // B: column l of tile

    v8f acc = {};
    for (int k0 = kBeg; k0 < kEnd; k0 += 64) {
#pragma unroll
        for (int u = 0; u < 2; ++u) {
            const int kk = k0 + u * 32;
            // A 16x32 bf16 layout: low lanes K in {0..7,16..23}, high lanes +8.
            const float* pa = xrow + kk + hi * 8;
            float4 a0 = *(const float4*)(pa);
            float4 a1 = *(const float4*)(pa + 4);
            float4 a2 = *(const float4*)(pa + 16);
            float4 a3 = *(const float4*)(pa + 20);
            // B 32x16 bf16 layout: low lanes K in {0..7,16..23}, high lanes +8.
            const float* pb = wrow + kk + hi * 8;
            float4 b0 = *(const float4*)(pb);
            float4 b1 = *(const float4*)(pb + 4);
            float4 b2 = *(const float4*)(pb + 16);
            float4 b3 = *(const float4*)(pb + 20);
            v16bf A = cvt16(a0, a1, a2, a3);
            v16bf B = cvt16(b0, b1, b2, b3);
            acc = __builtin_amdgcn_wmma_f32_16x16x32_bf16(
                false, A, false, B, (short)0, acc, false, false);
        }
        if (k0 + 64 < kEnd) __builtin_prefetch(wrow + k0 + 64 + hi * 16, 0, 1);
    }

    // Partial accumulators -> LDS (bank-conflict-free: lane fastest).
#pragma unroll
    for (int r = 0; r < 8; ++r) red[wid][r * 32 + lane] = acc[r];
    __syncthreads();

    // Each thread reduces one (r, lane) cell over the 8 waves, fixed order.
    const int t = threadIdx.x;             // t = r*32 + hi*16 + l
    float y = red[0][t];
#pragma unroll
    for (int w = 1; w < 8; ++w) y += red[w][t];

    const int r_  = t >> 5;
    const int hi_ = (t >> 4) & 1;
    const int n   = tn * 16 + (t & 15);
    const int m   = tm * 16 + hi_ * 8 + r_;
    y += bias[n];
    if (MODE == 1)      y = fmaxf(y, 0.0f);
    else if (MODE == 2) y = sigf(y);
    else if (MODE == 3) y = sigf(y) * aux[(size_t)m * Nout + n];
    Y[(size_t)m * Nout + n] = y;
}

// ---------------------------------------------------------------------------
// Sequential recurrence (tiny): usage/softmax over SLOTS, all T steps.
// Single block of 256 threads; LDS tree reductions for max & sum.
// ---------------------------------------------------------------------------
__global__ __launch_bounds__(256)
void wgt_scan_kernel(const float* __restrict__ logits, const float* __restrict__ usage0,
                     float* __restrict__ wgt_out) {
    __shared__ float red[NSLOTS];
    int   s     = threadIdx.x;
    float usage = usage0[s];
    for (int t = 0; t < TSTEPS; ++t) {
        float v = logits[t * NSLOTS + s] - 0.1f * usage;
        red[s] = v; __syncthreads();
        for (int off = NSLOTS / 2; off > 0; off >>= 1) {
            if (s < off) red[s] = fmaxf(red[s], red[s + off]);
            __syncthreads();
        }
        float mx = red[0]; __syncthreads();
        float e = expf(v - mx);
        red[s] = e; __syncthreads();
        for (int off = NSLOTS / 2; off > 0; off >>= 1) {
            if (s < off) red[s] += red[s + off];
            __syncthreads();
        }
        float w = e / red[0]; __syncthreads();
        wgt_out[t * NSLOTS + s] = w;
        usage += w;
    }
}

// ---------------------------------------------------------------------------
// Parallel memory replay: each thread owns 4 mem[s][d..d+3] cells (float4),
// replays 64 erase/add steps. grid (DIMX/1024, NSLOTS), block 256.
// erase/add (512 KB each) stay L2-resident across the 256 s-rows.
// ---------------------------------------------------------------------------
__global__ __launch_bounds__(256)
void mem_update_kernel(const float* __restrict__ mem0, const float* __restrict__ wgt,
                       const float* __restrict__ erase, const float* __restrict__ add,
                       float* __restrict__ out) {
    int s = blockIdx.y;
    int d = (blockIdx.x * blockDim.x + threadIdx.x) * 4;
    float4 m = *(const float4*)(mem0 + (size_t)s * DIMX + d);
#pragma unroll 4
    for (int t = 0; t < TSTEPS; ++t) {
        float  w = 0.5f * wgt[t * NSLOTS + s];      // scalar-uniform per block
        float4 e = *(const float4*)(erase + t * DIMX + d);
        float4 a = *(const float4*)(add   + t * DIMX + d);
        m.x = m.x * (1.0f - w * e.x) + w * a.x;
        m.y = m.y * (1.0f - w * e.y) + w * a.y;
        m.z = m.z * (1.0f - w * e.z) + w * a.z;
        m.w = m.w * (1.0f - w * e.w) + w * a.w;
    }
    *(float4*)(out + (size_t)s * DIMX + d) = m;
}

// ---------------------------------------------------------------------------
extern "C" void kernel_launch(void* const* d_in, const int* in_sizes, int n_in,
                              void* d_out, int out_size, void* d_ws, size_t ws_size,
                              hipStream_t stream) {
    const float* traces  = (const float*)d_in[0];
    const float* noise   = (const float*)d_in[1];
    const float* memory0 = (const float*)d_in[2];
    const float* usage0  = (const float*)d_in[3];
    const float* W1  = (const float*)d_in[4];
    const float* b1  = (const float*)d_in[5];
    const float* W2  = (const float*)d_in[6];
    const float* b2  = (const float*)d_in[7];
    const float* Wl  = (const float*)d_in[8];
    const float* bl  = (const float*)d_in[9];
    const float* vw  = (const float*)d_in[10];
    const float* vb  = (const float*)d_in[11];
    const float* Wg  = (const float*)d_in[12];
    const float* bg  = (const float*)d_in[13];
    const float* Wa  = (const float*)d_in[14];
    const float* ba  = (const float*)d_in[15];
    const float* We  = (const float*)d_in[16];
    const float* be  = (const float*)d_in[17];
    const float* Wad = (const float*)d_in[18];
    const float* bad = (const float*)d_in[19];
    const int*   idx = (const int*)d_in[20];
    float* out = (float*)d_out;

    // Workspace layout (floats)
    float* ws      = (float*)d_ws;
    const size_t TD = (size_t)TSTEPS * DIMX;     // 131072
    float* noisy   = ws;
    float* voltage = ws + 1 * TD;
    float* hbuf    = ws + 2 * TD;
    float* xbuf    = ws + 3 * TD;
    float* ligv    = ws + 4 * TD;
    float* gated   = ws + 5 * TD;
    float* erase_  = ws + 6 * TD;
    float* add_    = ws + 7 * TD;
    float* logits  = ws + 8 * TD;                       // 64*256
    float* wgtbuf  = logits + (size_t)TSTEPS * NSLOTS;  // 64*256

    dim3 blk(256);
    dim3 gElem(DIMX / 256, TSTEPS);

    prep_kernel<<<gElem, blk, 0, stream>>>(traces, noise, idx, vw, vb, noisy, voltage);

    const int gBig   = 4 * (DIMX / 16);    // 512 blocks (one per output tile)
    const int gSmall = 4 * (NSLOTS / 16);  // 64 blocks
    // h = relu(noisy @ W1^T + b1)
    gemm_wmma_bf16<1><<<gBig, blk, 0, stream>>>(noisy, W1, b1, nullptr, hbuf, DIMX, DIMX);
    // x = h @ W2^T + b2
    gemm_wmma_bf16<0><<<gBig, blk, 0, stream>>>(hbuf, W2, b2, nullptr, xbuf, DIMX, DIMX);
    // ligv = sigmoid(x @ Wl^T + bl) * voltage
    gemm_wmma_bf16<3><<<gBig, blk, 0, stream>>>(xbuf, Wl, bl, voltage, ligv, DIMX, DIMX);
    // gated = x * sigmoid(ligv @ Wg^T + bg)
    gemm_wmma_bf16<3><<<gBig, blk, 0, stream>>>(ligv, Wg, bg, xbuf, gated, DIMX, DIMX);
    // logits = gated @ Wa^T + ba
    gemm_wmma_bf16<0><<<gSmall, blk, 0, stream>>>(gated, Wa, ba, nullptr, logits, DIMX, NSLOTS);
    // erase = sigmoid(gated @ We^T + be)
    gemm_wmma_bf16<2><<<gBig, blk, 0, stream>>>(gated, We, be, nullptr, erase_, DIMX, DIMX);
    // add = gated @ Wad^T + bad
    gemm_wmma_bf16<0><<<gBig, blk, 0, stream>>>(gated, Wad, bad, nullptr, add_, DIMX, DIMX);

    // Serial softmax/usage recurrence (tiny), then parallel memory replay.
    wgt_scan_kernel<<<1, NSLOTS, 0, stream>>>(logits, usage0, wgtbuf);
    dim3 gMem(DIMX / 1024, NSLOTS);
    mem_update_kernel<<<gMem, blk, 0, stream>>>(memory0, wgtbuf, erase_, add_, out);
}
